// MultiheadAttention_25460566131051
// MI455X (gfx1250) — matrix operand order repaired and checked by reading, not compiled, not run on hardware
//
#include <hip/hip_runtime.h>
#include <stdint.h>

// ---------------------------------------------------------------------------
// Fused causal MHA for gfx1250 (MI455X).  B=4, S=2048, D=1024, H=16, hd=64.
// Compute-bound (~137 GFLOP vs ~80 MB HBM traffic) -> run everything through
// v_wmma_f32_16x16x32_bf16 (f32 accumulate).  Attention is computed
// transposed (S^T = K Q^T, O^T = V^T P^T): softmax stats are per-lane and the
// P^T B-fragment needs only one shfl_xor(16) half-wave exchange.
// Wave ids go through readfirstlane so all WMMA control flow is provably
// wave-uniform (EXEC all-ones, scalar branches).
// Workspace (~88 MB): xb | wq..wo bf16 | Q | K | V^T | Ctx (all bf16).
// ---------------------------------------------------------------------------

typedef __attribute__((ext_vector_type(16))) __bf16 v16bf;
typedef __attribute__((ext_vector_type(8)))  float  v8f;

union Frag16 { uint32_t u[8]; uint4 q[2]; v16bf v; };

__device__ __forceinline__ uint32_t f2bf(float f) {
  uint32_t u = __float_as_uint(f);
  return (u + 0x7FFFu + ((u >> 16) & 1u)) >> 16;   // RNE fp32 -> bf16
}
__device__ __forceinline__ uint32_t packbf(float a, float b) {
  return (f2bf(b) << 16) | f2bf(a);
}
__device__ __forceinline__ v8f wmma_bf16(const Frag16& a, const Frag16& b, v8f c) {
  return __builtin_amdgcn_wmma_f32_16x16x32_bf16(false, a.v, false, b.v,
                                                 (short)0, c, false, false);
}

static constexpr int DM = 1024;   // d_model
static constexpr int SQ = 2048;   // sequence
static constexpr int NH = 16;     // heads
static constexpr int HD = 64;     // head dim
static constexpr float NEG_BIG = -1.0e30f;

// ---------------------------------------------------------------- convert --
// Vectorized fp32 -> bf16 (4 elems/thread/iter).  n is a multiple of 4.
__global__ void cvt_f32_bf16(const float* __restrict__ src,
                             unsigned short* __restrict__ dst, int n4) {
  for (int i = blockIdx.x * blockDim.x + threadIdx.x; i < n4;
       i += gridDim.x * blockDim.x) {
    const float4 f = reinterpret_cast<const float4*>(src)[i];
    uint2 o;
    o.x = packbf(f.x, f.y);
    o.y = packbf(f.z, f.w);
    reinterpret_cast<uint2*>(dst)[i] = o;
  }
}

// ------------------------------------------------------------------- GEMM --
// Y = X[M,1024] @ W[N,1024]^T, per-wave tile 32(M) x 64(N), K-step 32,
// double-buffered (ping-pong) so next fragments are in flight during WMMAs.
// MODE 0: Q/K -> bf16 [b,h,s,64]   MODE 2: V^T -> bf16 [b,h,64,s]
// MODE 3: final -> fp32 [b,s,e]
struct KStep { Frag16 a[2]; Frag16 b[4]; };

__device__ __forceinline__ void load_step(KStep& s,
    const unsigned short* xr0, const unsigned short* xr1,
    const unsigned short* const* wr, int k0, int hi2) {
  const int ca = k0 + 8 * hi2;               // A-layout K offset
  s.a[0].q[0] = *reinterpret_cast<const uint4*>(xr0 + ca);
  s.a[0].q[1] = *reinterpret_cast<const uint4*>(xr0 + ca + 16);
  s.a[1].q[0] = *reinterpret_cast<const uint4*>(xr1 + ca);
  s.a[1].q[1] = *reinterpret_cast<const uint4*>(xr1 + ca + 16);
  const int cb = k0 + 8 * hi2;               // B-layout K offset
#pragma unroll
  for (int ni = 0; ni < 4; ++ni) {
    s.b[ni].q[0] = *reinterpret_cast<const uint4*>(wr[ni] + cb);
    s.b[ni].q[1] = *reinterpret_cast<const uint4*>(wr[ni] + cb + 16);
  }
}
__device__ __forceinline__ void mma_step(v8f acc[2][4], const KStep& s) {
#pragma unroll
  for (int ni = 0; ni < 4; ++ni) {
    acc[0][ni] = wmma_bf16(s.a[0], s.b[ni], acc[0][ni]);
    acc[1][ni] = wmma_bf16(s.a[1], s.b[ni], acc[1][ni]);
  }
}

template <int MODE>
__global__ __launch_bounds__(128) void gemm_nt_bf16(
    const unsigned short* __restrict__ X, const unsigned short* __restrict__ W,
    void* __restrict__ Y) {
  const int lane = threadIdx.x & 31;
  const int wv   = __builtin_amdgcn_readfirstlane(threadIdx.x >> 5);
  const int lo   = lane & 15;
  const int hi2  = lane >> 4;                 // 0 or 1
  const int m0   = blockIdx.x * 32;           // 256 m-tiles
  const int n0   = (blockIdx.y * 4 + wv) * 64;

  v8f acc[2][4];
#pragma unroll
  for (int a = 0; a < 2; ++a)
#pragma unroll
    for (int b = 0; b < 4; ++b)
#pragma unroll
      for (int j = 0; j < 8; ++j) acc[a][b][j] = 0.f;

  const unsigned short* xr0 = X + (size_t)(m0 + lo) * DM;
  const unsigned short* xr1 = xr0 + (size_t)16 * DM;
  const unsigned short* wr[4];
#pragma unroll
  for (int ni = 0; ni < 4; ++ni) wr[ni] = W + (size_t)(n0 + 16 * ni + lo) * DM;

  KStep s0, s1;
  load_step(s0, xr0, xr1, wr, 0, hi2);
  for (int k0 = 0; k0 < DM; k0 += 64) {       // 16 iterations, 2 k-steps each
    load_step(s1, xr0, xr1, wr, k0 + 32, hi2);
    mma_step(acc, s0);
    if (k0 + 64 < DM) load_step(s0, xr0, xr1, wr, k0 + 64, hi2);
    mma_step(acc, s1);
  }

#pragma unroll
  for (int mi = 0; mi < 2; ++mi) {
#pragma unroll
    for (int ni = 0; ni < 4; ++ni) {
      const int n     = n0 + 16 * ni + lo;
      const int mbase = m0 + 16 * mi + 8 * hi2;   // D-frag: M = mbase + j
      if (MODE == 3) {
        float* out = (float*)Y;
#pragma unroll
        for (int j = 0; j < 8; ++j)
          out[(size_t)(mbase + j) * DM + n] = acc[mi][ni][j];
      } else {
        const int bb = mbase >> 11;     // / 2048
        const int sb = mbase & 2047;
        const int hh = n >> 6;
        const int hd = n & 63;
        unsigned short* out = (unsigned short*)Y;
        if (MODE == 2) {                // V^T: 8 consecutive s positions
          uint4 w;
          w.x = packbf(acc[mi][ni][0], acc[mi][ni][1]);
          w.y = packbf(acc[mi][ni][2], acc[mi][ni][3]);
          w.z = packbf(acc[mi][ni][4], acc[mi][ni][5]);
          w.w = packbf(acc[mi][ni][6], acc[mi][ni][7]);
          *reinterpret_cast<uint4*>(
              out + (((size_t)(bb * NH + hh) * HD + hd) * SQ + sb)) = w;
        } else {                        // Q/K: [b,h,s,64]
#pragma unroll
          for (int j = 0; j < 8; ++j)
            out[((size_t)(bb * NH + hh) * SQ + (sb + j)) * HD + hd] =
                (unsigned short)f2bf(acc[mi][ni][j]);
        }
      }
    }
  }
}

// -------------------------------------------------------------- attention --
// One wave per (b, h, 16-query tile); flash-style causal softmax over 32-wide
// KV blocks.  All K and V fragments for a block are issued as one load clause
// at the top of the iteration; the softmax VALU work hides the V latency.
__global__ __launch_bounds__(128) void attn_kernel(
    const unsigned short* __restrict__ Q, const unsigned short* __restrict__ K,
    const unsigned short* __restrict__ VT, unsigned short* __restrict__ Ctx) {
  const int lane = threadIdx.x & 31;
  const int wv   = __builtin_amdgcn_readfirstlane(threadIdx.x >> 5);
  const int lo   = lane & 15;
  const int hi2  = lane >> 4;
  const int qt = blockIdx.x * 4 + wv;     // 0..127  (wave-uniform)
  const int h  = blockIdx.y;
  const int b  = blockIdx.z;
  const int q0 = qt * 16;

  const unsigned short* qp = Q  + (size_t)(b * NH + h) * SQ * HD;
  const unsigned short* kp = K  + (size_t)(b * NH + h) * SQ * HD;
  const unsigned short* vp = VT + (size_t)(b * NH + h) * HD * SQ;

  // Q^T B-fragments for the two d k-steps (d 0..31 / 32..63), loaded once.
  Frag16 qf[2];
  {
    const unsigned short* qr = qp + (size_t)(q0 + lo) * HD;
#pragma unroll
    for (int ks = 0; ks < 2; ++ks) {
      const int col = 32 * ks + 8 * hi2;
      qf[ks].q[0] = *reinterpret_cast<const uint4*>(qr + col);
      qf[ks].q[1] = *reinterpret_cast<const uint4*>(qr + col + 16);
    }
  }

  v8f acc[4];                             // O^T tiles: d 0-15,16-31,32-47,48-63
#pragma unroll
  for (int t = 0; t < 4; ++t)
#pragma unroll
    for (int j = 0; j < 8; ++j) acc[t][j] = 0.f;
  float m_run = NEG_BIG, l_run = 0.f;

  const int nblocks = ((q0 + 15) >> 5) + 1;   // wave-uniform trip count
  for (int jb = 0; jb < nblocks; ++jb) {
    const int kb = jb * 32;

    // ---- issue ALL loads for this block up front (K then V) --------------
    Frag16 kf[2][2];                      // [kv-tile][d-kstep]
#pragma unroll
    for (int t2 = 0; t2 < 2; ++t2) {
      const unsigned short* kr = kp + (size_t)(kb + 16 * t2 + lo) * HD;
#pragma unroll
      for (int ks = 0; ks < 2; ++ks) {
        const int col = 32 * ks + 8 * hi2;
        kf[t2][ks].q[0] = *reinterpret_cast<const uint4*>(kr + col);
        kf[t2][ks].q[1] = *reinterpret_cast<const uint4*>(kr + col + 16);
      }
    }
    Frag16 vf[4];                         // V^T A-fragments, used after softmax
#pragma unroll
    for (int t = 0; t < 4; ++t) {
      const unsigned short* vr = vp + (size_t)(16 * t + lo) * SQ + kb + 16 * hi2;
      vf[t].q[0] = *reinterpret_cast<const uint4*>(vr);
      vf[t].q[1] = *reinterpret_cast<const uint4*>(vr + 8);
    }

    // ---- S^T = K_block * Q^T  (two 16(kv) x 16(q) tiles) -----------------
    v8f st[2];
#pragma unroll
    for (int t2 = 0; t2 < 2; ++t2) {
#pragma unroll
      for (int j = 0; j < 8; ++j) st[t2][j] = 0.f;
      st[t2] = wmma_bf16(kf[t2][0], qf[0], st[t2]);
      st[t2] = wmma_bf16(kf[t2][1], qf[1], st[t2]);
    }
    float s[16];
#pragma unroll
    for (int j = 0; j < 8; ++j) {
      s[j]     = st[0][j] * 0.125f;       // 1/sqrt(64)
      s[8 + j] = st[1][j] * 0.125f;
    }
    if (kb + 31 > q0) {                   // diagonal block: causal mask
      const int qcol = q0 + lo;
#pragma unroll
      for (int j = 0; j < 8; ++j) {
        const int kv = kb + j + 8 * hi2;
        if (kv > qcol)      s[j]     = NEG_BIG;
        if (kv + 16 > qcol) s[8 + j] = NEG_BIG;
      }
    }
    // ---- online softmax (per-lane; combine lane pair l <-> l^16) ---------
    float mloc = s[0];
#pragma unroll
    for (int i = 1; i < 16; ++i) mloc = fmaxf(mloc, s[i]);
    mloc = fmaxf(mloc, __shfl_xor(mloc, 16, 32));
    const float m_new = fmaxf(m_run, mloc);
    const float scale = __expf(m_run - m_new);
    float p[16], psum = 0.f;
#pragma unroll
    for (int i = 0; i < 16; ++i) { p[i] = __expf(s[i] - m_new); psum += p[i]; }
    psum += __shfl_xor(psum, 16, 32);
    l_run = l_run * scale + psum;
    m_run = m_new;
#pragma unroll
    for (int t = 0; t < 4; ++t)
#pragma unroll
      for (int j = 0; j < 8; ++j) acc[t][j] *= scale;
    // ---- build P^T B-fragment (32kv x 16q): half-wave exchange only ------
    uint32_t p0[4], p1[4];
#pragma unroll
    for (int i = 0; i < 4; ++i) {
      p0[i] = packbf(p[2 * i], p[2 * i + 1]);
      p1[i] = packbf(p[8 + 2 * i], p[9 + 2 * i]);
    }
    Frag16 pf;
#pragma unroll
    for (int i = 0; i < 4; ++i) {
      const uint32_t x0 = (uint32_t)__shfl_xor((int)p0[i], 16, 32);
      const uint32_t x1 = (uint32_t)__shfl_xor((int)p1[i], 16, 32);
      pf.u[i]     = hi2 ? x1 : p0[i];     // K 0..7  | 16..23
      pf.u[4 + i] = hi2 ? p1[i] : x0;     // K 8..15 | 24..31
    }
    // ---- O^T += V^T_block * P^T  (V loads issued long ago) ---------------
#pragma unroll
    for (int t = 0; t < 4; ++t) acc[t] = wmma_bf16(vf[t], pf, acc[t]);
  }

  // ---- epilogue: Ctx[b, q, h*64 + d] = acc^T[d, q] / l  (packed b128) ----
  const float inv = 1.0f / l_run;
  unsigned short* cp =
      Ctx + ((size_t)(b * SQ + q0 + lo) * DM) + h * HD + 8 * hi2;
#pragma unroll
  for (int t = 0; t < 4; ++t) {
    uint4 w;
    w.x = packbf(acc[t][0] * inv, acc[t][1] * inv);
    w.y = packbf(acc[t][2] * inv, acc[t][3] * inv);
    w.z = packbf(acc[t][4] * inv, acc[t][5] * inv);
    w.w = packbf(acc[t][6] * inv, acc[t][7] * inv);
    *reinterpret_cast<uint4*>(cp + 16 * t) = w;
  }
}

// ------------------------------------------------------------------ launch --
extern "C" void kernel_launch(void* const* d_in, const int* in_sizes, int n_in,
                              void* d_out, int out_size, void* d_ws,
                              size_t ws_size, hipStream_t stream) {
  const float* x  = (const float*)d_in[0];
  const float* Wq = (const float*)d_in[1];
  const float* Wk = (const float*)d_in[2];
  const float* Wv = (const float*)d_in[3];
  const float* Wo = (const float*)d_in[4];

  const size_t NX = (size_t)4 * SQ * DM;   // 8,388,608 activations
  const size_t NW = (size_t)DM * DM;       // 1,048,576 per weight
  char* ws = (char*)d_ws;
  size_t o = 0;
  unsigned short* xb  = (unsigned short*)(ws + o); o += NX * 2;
  unsigned short* wqb = (unsigned short*)(ws + o); o += NW * 2;
  unsigned short* wkb = (unsigned short*)(ws + o); o += NW * 2;
  unsigned short* wvb = (unsigned short*)(ws + o); o += NW * 2;
  unsigned short* wob = (unsigned short*)(ws + o); o += NW * 2;
  unsigned short* Qb  = (unsigned short*)(ws + o); o += NX * 2;
  unsigned short* Kb  = (unsigned short*)(ws + o); o += NX * 2;
  unsigned short* VTb = (unsigned short*)(ws + o); o += NX * 2;
  unsigned short* Ctx = (unsigned short*)(ws + o); o += NX * 2;
  if (ws_size < o) return;                 // ~88 MB scratch required

  cvt_f32_bf16<<<1024, 256, 0, stream>>>(x, xb, (int)(NX / 4));
  cvt_f32_bf16<<<256, 256, 0, stream>>>(Wq, wqb, (int)(NW / 4));
  cvt_f32_bf16<<<256, 256, 0, stream>>>(Wk, wkb, (int)(NW / 4));
  cvt_f32_bf16<<<256, 256, 0, stream>>>(Wv, wvb, (int)(NW / 4));
  cvt_f32_bf16<<<256, 256, 0, stream>>>(Wo, wob, (int)(NW / 4));

  const dim3 gg(256, 4, 1), bb(128);
  gemm_nt_bf16<0><<<gg, bb, 0, stream>>>(xb, wqb, Qb);
  gemm_nt_bf16<0><<<gg, bb, 0, stream>>>(xb, wkb, Kb);
  gemm_nt_bf16<2><<<gg, bb, 0, stream>>>(xb, wvb, VTb);

  attn_kernel<<<dim3(32, NH, 4), 128, 0, stream>>>(Qb, Kb, VTb, Ctx);

  gemm_nt_bf16<3><<<gg, bb, 0, stream>>>(Ctx, wob, d_out);
}
